// ConvolutionalMultiheadAttention_3633542332460
// MI455X (gfx1250) — matrix operand order repaired and checked by reading, not compiled, not run on hardware
//
#include <hip/hip_runtime.h>
#include <hip/hip_bf16.h>

typedef __bf16 v16bf __attribute__((ext_vector_type(16)));
typedef __bf16 v8bf  __attribute__((ext_vector_type(8)));
typedef float  v8f   __attribute__((ext_vector_type(8)));

#define C_CH 512
#define SPA  1024   // 32*32
#define NB   2      // batch
#define NH   8
#define DH   64
#define NBH  16     // NB*NH
#define PW   34     // padded spatial width (32 + halo)
#define PSPA (PW * PW)          // 1156 padded positions per image

// ---------------------------------------------------------------------------
// Packing kernels: f32 -> bf16 into GEMM-friendly layouts
// ---------------------------------------------------------------------------
// Padded NHWC: xp[n][yy][xx][ci], yy/xx in [0,34), zero halo at border.
__global__ void pack_x_pad(const float* __restrict__ x, __bf16* __restrict__ xp) {
  int i = blockIdx.x * blockDim.x + threadIdx.x;     // over NB*34*34*512
  if (i >= NB * PSPA * C_CH) return;
  int ci = i & 511;
  int r  = i >> 9;
  int xx = r % PW;
  int r2 = r / PW;
  int yy = r2 % PW;
  int n  = r2 / PW;
  float v = 0.0f;
  if (yy >= 1 && yy <= 32 && xx >= 1 && xx <= 32) {
    int s = (yy - 1) * 32 + (xx - 1);
    v = x[((size_t)n * C_CH + ci) * SPA + s];
  }
  xp[i] = (__bf16)v;
}

__global__ void pack_w_tapco(const float* __restrict__ w, __bf16* __restrict__ wp) {
  int i = blockIdx.x * blockDim.x + threadIdx.x;     // over 9*512*512
  if (i >= 9 * C_CH * C_CH) return;
  int ci  = i & 511;
  int co  = (i >> 9) & 511;
  int tap = i >> 18;
  // w layout: [co][ci][ky][kx] ; tap = ky*3+kx
  wp[i] = (__bf16)w[((size_t)co * C_CH + ci) * 9 + tap];
}

__global__ void fill_zero_bf16(__bf16* __restrict__ p, int n) {
  int i = blockIdx.x * blockDim.x + threadIdx.x;
  if (i < n) p[i] = (__bf16)0.0f;
}

// ---------------------------------------------------------------------------
// 3x3 conv as implicit GEMM (M=co 16, N=pos 64, K=9*512) with WMMA bf16.
// Input is padded NHWC -> every tap load is in-bounds (no predication).
// mode 0: rotary -> q[bh][s][d]     mode 1: rotary -> k[bh][s][d]
// mode 2: -> vT[bh*64+d][s]         mode 3: -> f32 NCHW out
// ---------------------------------------------------------------------------
__global__ __launch_bounds__(128) void conv3x3_wmma(
    const __bf16* __restrict__ in,    // [n][34][34][ci] bf16, zero halo
    const __bf16* __restrict__ wp,    // [9][co][ci] bf16
    const float*  __restrict__ bias,  // [co]
    __bf16* __restrict__ out_bf,
    float*  __restrict__ out_f32,
    int mode)
{
  const int wave = (blockIdx.x * blockDim.x + threadIdx.x) >> 5;
  const int lane = threadIdx.x & 31;
  const int half = lane >> 4;
  const int lm   = lane & 15;
  const int co_tile = wave & 31;        // 32 tiles of 16 channels
  const int posblk  = wave >> 5;        // 32 blocks of 64 positions
  const int co_base = co_tile * 16;
  const int pos0    = posblk * 64;

  v8f acc[4];
#pragma unroll
  for (int t = 0; t < 4; ++t)
#pragma unroll
    for (int i = 0; i < 8; ++i) acc[t][i] = 0.0f;

  // Loop-invariant per-lane "center" pointers (tap dy=dx=0) into padded input.
  const __bf16* bcen[4];
#pragma unroll
  for (int t = 0; t < 4; ++t) {
    int p = pos0 + 16 * t + lm;
    int n = p >> 10;
    int s = p & 1023;
    int y = s >> 5, xcol = s & 31;
    bcen[t] = in + ((size_t)(n * PW + (y + 1)) * PW + (xcol + 1)) * C_CH + half * 16;
  }

  // A fragment: lane holds row M=lm, K = half*16 + {0..15}
  const __bf16* arow0 = wp + (size_t)(co_base + lm) * C_CH + half * 16;

  for (int tap = 0; tap < 9; ++tap) {
    const int dy = tap / 3 - 1, dx = tap % 3 - 1;
    const int boff = (dy * PW + dx) * C_CH;           // element offset per tap
    const __bf16* ar = arow0 + (size_t)tap * C_CH * C_CH;
    const __bf16* bp0 = bcen[0] + boff;
    const __bf16* bp1 = bcen[1] + boff;
    const __bf16* bp2 = bcen[2] + boff;
    const __bf16* bp3 = bcen[3] + boff;
#pragma unroll 4
    for (int k0 = 0; k0 < C_CH; k0 += 32) {
      v16bf A;
      ((v8bf*)&A)[0] = *(const v8bf*)(ar + k0);
      ((v8bf*)&A)[1] = *(const v8bf*)(ar + k0 + 8);
      v16bf B0, B1, B2, B3;
      ((v8bf*)&B0)[0] = *(const v8bf*)(bp0 + k0);
      ((v8bf*)&B0)[1] = *(const v8bf*)(bp0 + k0 + 8);
      ((v8bf*)&B1)[0] = *(const v8bf*)(bp1 + k0);
      ((v8bf*)&B1)[1] = *(const v8bf*)(bp1 + k0 + 8);
      ((v8bf*)&B2)[0] = *(const v8bf*)(bp2 + k0);
      ((v8bf*)&B2)[1] = *(const v8bf*)(bp2 + k0 + 8);
      ((v8bf*)&B3)[0] = *(const v8bf*)(bp3 + k0);
      ((v8bf*)&B3)[1] = *(const v8bf*)(bp3 + k0 + 8);
      acc[0] = __builtin_amdgcn_wmma_f32_16x16x32_bf16(false, A, false, B0, (short)0, acc[0], false, false);
      acc[1] = __builtin_amdgcn_wmma_f32_16x16x32_bf16(false, A, false, B1, (short)0, acc[1], false, false);
      acc[2] = __builtin_amdgcn_wmma_f32_16x16x32_bf16(false, A, false, B2, (short)0, acc[2], false, false);
      acc[3] = __builtin_amdgcn_wmma_f32_16x16x32_bf16(false, A, false, B3, (short)0, acc[3], false, false);
    }
  }

  // Epilogue. D layout: lane L -> N = L%16 (position), VGPR r -> M = r + 8*(L/16)
#pragma unroll
  for (int t = 0; t < 4; ++t) {
    int p = pos0 + 16 * t + lm;
    int n = p >> 10;
    int s = p & 1023;
    if (mode <= 1) {
      // rotary on channel pairs (co even, co+1); ln(10000)/32 = 0.287823136...
#pragma unroll
      for (int u = 0; u < 4; ++u) {
        int co0 = co_base + 2 * u + 8 * half;           // even
        float v0 = acc[t][2 * u]     + bias[co0];
        float v1 = acc[t][2 * u + 1] + bias[co0 + 1];
        int d0 = co0 & 63;
        int pi = d0 >> 1;
        float ang = (float)s * __expf(-(float)pi * 0.28782313662425574f);
        float sn, cs;
        __sincosf(ang, &sn, &cs);
        float r0 = v0 * cs - v1 * sn;
        float r1 = v1 * cs + v0 * sn;
        int h = co0 >> 6;
        size_t idx = ((size_t)((n * NH + h) * SPA + s)) * DH + d0;
        out_bf[idx]     = (__bf16)r0;
        out_bf[idx + 1] = (__bf16)r1;
      }
    } else if (mode == 2) {
#pragma unroll
      for (int r = 0; r < 8; ++r) {
        int co = co_base + r + 8 * half;
        float v = acc[t][r] + bias[co];
        // vT[(n*8+h)*64+d][s] == [(n*512+co)][s]
        out_bf[(size_t)(n * C_CH + co) * SPA + s] = (__bf16)v;
      }
    } else {
#pragma unroll
      for (int r = 0; r < 8; ++r) {
        int co = co_base + r + 8 * half;
        out_f32[(size_t)(n * C_CH + co) * SPA + s] = acc[t][r] + bias[co];
      }
    }
  }
}

// ---------------------------------------------------------------------------
// qk = (Q K^T) / sqrt(512), per (b,h). f32 out straight into d_out qk region.
// ---------------------------------------------------------------------------
__global__ __launch_bounds__(128) void qk_wmma(
    const __bf16* __restrict__ q,   // [bh][s][64]
    const __bf16* __restrict__ k,   // [bh][s][64]
    float* __restrict__ qk)         // [bh][sq][sk]
{
  const int wave = (blockIdx.x * blockDim.x + threadIdx.x) >> 5;
  const int lane = threadIdx.x & 31;
  const int half = lane >> 4, lm = lane & 15;
  const int skb = wave & 15;            // 16 blocks of 64 keys
  const int sqt = (wave >> 4) & 63;     // 64 query tiles
  const int bh  = wave >> 10;           // 16

  v8f acc[4];
#pragma unroll
  for (int t = 0; t < 4; ++t)
#pragma unroll
    for (int i = 0; i < 8; ++i) acc[t][i] = 0.0f;

  const __bf16* ar = q + (size_t)(bh * SPA + sqt * 16 + lm) * DH + half * 16;
  const __bf16* br[4];
#pragma unroll
  for (int t = 0; t < 4; ++t)
    br[t] = k + (size_t)(bh * SPA + skb * 64 + 16 * t + lm) * DH + half * 16;

#pragma unroll
  for (int k0 = 0; k0 < DH; k0 += 32) {
    v16bf A;
    ((v8bf*)&A)[0] = *(const v8bf*)(ar + k0);
    ((v8bf*)&A)[1] = *(const v8bf*)(ar + k0 + 8);
#pragma unroll
    for (int t = 0; t < 4; ++t) {
      v16bf B;
      ((v8bf*)&B)[0] = *(const v8bf*)(br[t] + k0);
      ((v8bf*)&B)[1] = *(const v8bf*)(br[t] + k0 + 8);
      acc[t] = __builtin_amdgcn_wmma_f32_16x16x32_bf16(
          false, A, false, B, (short)0, acc[t], false, false);
    }
  }

  const float scale = 0.044194173824159216f;  // 1/sqrt(512)
#pragma unroll
  for (int t = 0; t < 4; ++t)
#pragma unroll
    for (int r = 0; r < 8; ++r) {
      int sq = sqt * 16 + r + 8 * half;
      int sk = skb * 64 + 16 * t + lm;
      qk[((size_t)bh * SPA + sq) * SPA + sk] = acc[t][r] * scale;
    }
}

// ---------------------------------------------------------------------------
// Row softmax over the stored qk logits -> bf16 P
// ---------------------------------------------------------------------------
__global__ __launch_bounds__(256) void softmax_rows(
    const float* __restrict__ qk, __bf16* __restrict__ P)
{
  const int row = blockIdx.x;           // 16*1024 rows
  const int tid = threadIdx.x;
  const float* rp = qk + (size_t)row * SPA;
  __shared__ float red[256];

  float v[4];
  float m = -3.4e38f;
#pragma unroll
  for (int t = 0; t < 4; ++t) { v[t] = rp[tid + 256 * t]; m = fmaxf(m, v[t]); }
  red[tid] = m; __syncthreads();
  for (int s = 128; s > 0; s >>= 1) {
    if (tid < s) red[tid] = fmaxf(red[tid], red[tid + s]);
    __syncthreads();
  }
  m = red[0]; __syncthreads();

  float e[4], ssum = 0.0f;
#pragma unroll
  for (int t = 0; t < 4; ++t) { e[t] = __expf(v[t] - m); ssum += e[t]; }
  red[tid] = ssum; __syncthreads();
  for (int s = 128; s > 0; s >>= 1) {
    if (tid < s) red[tid] += red[tid + s];
    __syncthreads();
  }
  float inv = 1.0f / red[0];
#pragma unroll
  for (int t = 0; t < 4; ++t)
    P[(size_t)row * SPA + tid + 256 * t] = (__bf16)(e[t] * inv);
}

// ---------------------------------------------------------------------------
// a = P V : per (b,h), M=sq, N=d(64), K=sk(1024). Writes padded NHWC for conv_o.
// ---------------------------------------------------------------------------
__global__ __launch_bounds__(128) void pv_wmma(
    const __bf16* __restrict__ P,    // [bh][sq][sk]
    const __bf16* __restrict__ vt,   // [bh*64+d][s]
    __bf16* __restrict__ ap)         // [n][34][34][c], halo pre-zeroed
{
  const int wave = (blockIdx.x * blockDim.x + threadIdx.x) >> 5;
  const int lane = threadIdx.x & 31;
  const int half = lane >> 4, lm = lane & 15;
  const int sqt = wave & 63;
  const int bh  = wave >> 6;            // 16

  v8f acc[4];
#pragma unroll
  for (int t = 0; t < 4; ++t)
#pragma unroll
    for (int i = 0; i < 8; ++i) acc[t][i] = 0.0f;

  const __bf16* ar = P + (size_t)(bh * SPA + sqt * 16 + lm) * SPA + half * 16;
  const __bf16* br[4];
#pragma unroll
  for (int t = 0; t < 4; ++t)
    br[t] = vt + (size_t)(bh * DH + 16 * t + lm) * SPA + half * 16;

#pragma unroll 4
  for (int k0 = 0; k0 < SPA; k0 += 32) {
    v16bf A;
    ((v8bf*)&A)[0] = *(const v8bf*)(ar + k0);
    ((v8bf*)&A)[1] = *(const v8bf*)(ar + k0 + 8);
#pragma unroll
    for (int t = 0; t < 4; ++t) {
      v16bf B;
      ((v8bf*)&B)[0] = *(const v8bf*)(br[t] + k0);
      ((v8bf*)&B)[1] = *(const v8bf*)(br[t] + k0 + 8);
      acc[t] = __builtin_amdgcn_wmma_f32_16x16x32_bf16(
          false, A, false, B, (short)0, acc[t], false, false);
    }
  }

  const int n = bh >> 3, h = bh & 7;
#pragma unroll
  for (int t = 0; t < 4; ++t)
#pragma unroll
    for (int r = 0; r < 8; ++r) {
      int sq = sqt * 16 + r + 8 * half;
      int d  = 16 * t + lm;
      int y = sq >> 5, xcol = sq & 31;
      ap[((size_t)(n * PW + (y + 1)) * PW + (xcol + 1)) * C_CH + h * DH + d] =
          (__bf16)acc[t][r];
    }
}

// ---------------------------------------------------------------------------
// Launcher
// ---------------------------------------------------------------------------
extern "C" void kernel_launch(void* const* d_in, const int* in_sizes, int n_in,
                              void* d_out, int out_size, void* d_ws, size_t ws_size,
                              hipStream_t stream) {
  (void)in_sizes; (void)n_in; (void)out_size; (void)ws_size;
  const float* x  = (const float*)d_in[0];
  const float* wq = (const float*)d_in[1];
  const float* bq = (const float*)d_in[2];
  const float* wk = (const float*)d_in[3];
  const float* bk = (const float*)d_in[4];
  const float* wv = (const float*)d_in[5];
  const float* bv = (const float*)d_in[6];
  const float* wo = (const float*)d_in[7];
  const float* bo = (const float*)d_in[8];

  char* ws = (char*)d_ws;
  size_t off = 0;
  auto carve = [&](size_t bytes) -> char* {
    char* p = ws + off;
    off += (bytes + 255) & ~(size_t)255;
    return p;
  };
  const size_t XPAD_E = (size_t)NB * PSPA * C_CH;  // 1,183,744 (padded NHWC)
  const size_t WP_E   = (size_t)9 * C_CH * C_CH;   // 2,359,296
  const size_t QK_E   = (size_t)NBH * SPA * DH;    // 1,048,576
  const size_t P_E    = (size_t)NBH * SPA * SPA;   // 16,777,216

  __bf16* xp  = (__bf16*)carve(XPAD_E * 2);
  __bf16* wqp = (__bf16*)carve(WP_E * 2);
  __bf16* wkp = (__bf16*)carve(WP_E * 2);
  __bf16* wvp = (__bf16*)carve(WP_E * 2);
  __bf16* wop = (__bf16*)carve(WP_E * 2);
  __bf16* qb  = (__bf16*)carve(QK_E * 2);
  __bf16* kb  = (__bf16*)carve(QK_E * 2);
  __bf16* vtb = (__bf16*)carve(QK_E * 2);
  __bf16* pb  = (__bf16*)carve(P_E * 2);
  __bf16* apb = (__bf16*)carve(XPAD_E * 2);

  float* outp = (float*)d_out;                               // (2,512,32,32)
  float* qkp  = (float*)d_out + (size_t)NB * C_CH * SPA;     // (2,8,1024,1024)

  pack_x_pad<<<(XPAD_E + 255) / 256, 256, 0, stream>>>(x, xp);
  pack_w_tapco<<<(WP_E + 255) / 256, 256, 0, stream>>>(wq, wqp);
  pack_w_tapco<<<(WP_E + 255) / 256, 256, 0, stream>>>(wk, wkp);
  pack_w_tapco<<<(WP_E + 255) / 256, 256, 0, stream>>>(wv, wvp);
  pack_w_tapco<<<(WP_E + 255) / 256, 256, 0, stream>>>(wo, wop);
  fill_zero_bf16<<<(XPAD_E + 255) / 256, 256, 0, stream>>>(apb, (int)XPAD_E);

  // 1024 wave-tasks, 4 waves/block
  conv3x3_wmma<<<256, 128, 0, stream>>>(xp, wqp, bq, qb,  nullptr, 0);
  conv3x3_wmma<<<256, 128, 0, stream>>>(xp, wkp, bk, kb,  nullptr, 1);
  conv3x3_wmma<<<256, 128, 0, stream>>>(xp, wvp, bv, vtb, nullptr, 2);

  qk_wmma<<<4096, 128, 0, stream>>>(qb, kb, qkp);            // 16384 wave-tasks
  softmax_rows<<<NBH * SPA, 256, 0, stream>>>(qkp, pb);
  pv_wmma<<<256, 128, 0, stream>>>(pb, vtb, apb);            // 1024 wave-tasks

  conv3x3_wmma<<<256, 128, 0, stream>>>(apb, wop, bo, nullptr, outp, 3);
}
